// WeightedEdgeConv_38345468018707
// MI455X (gfx1250) — matrix operand order repaired and checked by reading, not compiled, not run on hardware
//
#include <hip/hip_runtime.h>
#include <math.h>

#define NN 100000
#define NE 1600000
#define DIM 64
#define EDGE_GROUPS (NE / 16)       // 100000 groups of 16 edges
#define EDGE_BLOCKS 1250            // 10000 waves -> exactly 10 groups/wave
#define EDGE_WAVES (EDGE_BLOCKS * 8)

typedef __attribute__((ext_vector_type(16))) _Float16 v16h;
typedef __attribute__((ext_vector_type(8)))  float    v8f;

// ---- A fragment: 16x32 f16, gathered from a per-lane fp32 feature row ----
// ISA layout (16-bit A 16x32): lane m=lane&15; lanes<16 kh=0, lanes>=16 kh=8;
// VGPR r<4 -> K = kh+2r, kh+2r+1 ; VGPR r>=4 -> K = 16+kh+2(r-4), +1
__device__ __forceinline__ v16h load_a_frag(const float* __restrict__ row,
                                            int kbase, int lane) {
  const int kh = (lane >> 4) << 3;
  v16h a;
#pragma unroll
  for (int r = 0; r < 8; ++r) {
    const int k = kbase + ((r < 4) ? (kh + 2 * r) : (16 + kh + 2 * (r - 4)));
    const float2 f = *(const float2*)(row + k);
    a[2 * r]     = (_Float16)f.x;
    a[2 * r + 1] = (_Float16)f.y;
  }
  return a;
}

// ---- B fragment: 32x16 f16 from LDS, weights stored transposed w[n][k] ----
// ISA layout: lane n=lane&15; kh=8*(lane>=16); elems 0..7 K=kh..kh+7, elems 8..15 K=16+kh..16+kh+7.
__device__ __forceinline__ v16h load_b_frag(const _Float16* __restrict__ w,
                                            int stride, int n0, int k0, int lane) {
  const _Float16* p = w + (n0 + (lane & 15)) * stride + k0 + ((lane >> 4) << 3);
  v16h b;
#pragma unroll
  for (int r = 0; r < 8; ++r) { const int o = (r < 4) ? 2 * r : 2 * r + 8; b[2 * r] = p[o]; b[2 * r + 1] = p[o + 1]; }
  return b;
}

// Native CDNA5 fp32 L2 atomic add (no-return form, tracked by STOREcnt;
// S_ENDPGM performs an implicit wait-idle so completion is guaranteed).
__device__ __forceinline__ void atomic_add_f32(float* p, float v) {
  asm volatile("global_atomic_add_f32 %0, %1, off" :: "v"(p), "v"(v) : "memory");
}

// ------------------- zero the accumulation region -------------------
__global__ __launch_bounds__(256) void wec_zero_kernel(float4* __restrict__ p,
                                                       long long n4) {
  const long long i = (long long)blockIdx.x * blockDim.x + threadIdx.x;
  if (i < n4) p[i] = make_float4(0.f, 0.f, 0.f, 0.f);
}

// ------------------- x_t = x @ wt + bt  via WMMA -------------------
#define WT_STRIDE 68
__global__ __launch_bounds__(256)
void wec_node_transform_kernel(const float* __restrict__ x,
                               const float* __restrict__ wt,
                               const float* __restrict__ bt,
                               float* __restrict__ xt) {
  __shared__ _Float16 wtt[DIM][WT_STRIDE];   // wt transposed: [n][k], f16
  __shared__ float bts[DIM];
  const int tid = threadIdx.x, wv = tid >> 5, lane = tid & 31;

  for (int i = tid; i < DIM * DIM; i += 256) {
    const int n = i & 63, k = i >> 6;
    wtt[n][k] = (_Float16)wt[k * DIM + n];
  }
  if (tid < DIM) bts[tid] = bt[tid];
  __syncthreads();

  const long long group = (long long)blockIdx.x * 8 + wv;   // 16 rows per wave
  if (group >= (NN / 16)) return;                           // wave-uniform exit
  const long long base = group * 16;
  const int m16 = lane & 15;
  const float* xr = x + (base + m16) * DIM;

  const v16h a0 = load_a_frag(xr, 0, lane);
  const v16h a1 = load_a_frag(xr, 32, lane);

#pragma unroll
  for (int t = 0; t < 4; ++t) {
    const int n = t * 16 + m16;
    v8f acc;
#pragma unroll
    for (int r = 0; r < 8; ++r) acc[r] = bts[n];
    v16h b;
    b = load_b_frag(&wtt[0][0], WT_STRIDE, t * 16, 0, lane);
    acc = __builtin_amdgcn_wmma_f32_16x16x32_f16(false, a0, false, b, (short)0, acc, false, false);
    b = load_b_frag(&wtt[0][0], WT_STRIDE, t * 16, 32, lane);
    acc = __builtin_amdgcn_wmma_f32_16x16x32_f16(false, a1, false, b, (short)0, acc, false, false);
#pragma unroll
    for (int r = 0; r < 8; ++r)  // C layout: M = r + 8*(lane>=16), N = n
      xt[(base + r + ((lane >> 4) << 3)) * DIM + n] = acc[r];
  }
}

// ------- edge MLP (WMMA) + sigmoid + weighted scatter-add (fused) -------
// Each wave processes 10 groups of 16 edges; per-edge scalars are exchanged
// via wave shuffles so the group loop contains no block barriers.
#define W1_STRIDE 132
__global__ __launch_bounds__(256)
void wec_edge_mlp_scatter_kernel(const float* __restrict__ x,
                                 const long long* __restrict__ ei,
                                 const float* __restrict__ pos,
                                 const float* __restrict__ w1,
                                 const float* __restrict__ b1,
                                 const float* __restrict__ w2,
                                 const float* __restrict__ b2,
                                 const float* __restrict__ xt,
                                 float* __restrict__ out,
                                 float* __restrict__ ew_out) {
  __shared__ _Float16 w1t[DIM][W1_STRIDE];   // w1 rows 0..127, transposed [n][k]
  __shared__ float b1s[DIM], w2s[DIM], wls[DIM];

  const int tid = threadIdx.x, wv = tid >> 5, lane = tid & 31;
  const int m16 = lane & 15, half = lane >> 4;

  // stage MLP weights as f16 (transposed for contiguous B-fragment loads)
  for (int i = tid; i < DIM * 128; i += 256) {
    const int n = i & 63, k = i >> 6;
    w1t[n][k] = (_Float16)w1[k * DIM + n];
  }
  if (tid < DIM) {
    b1s[tid] = b1[tid];
    w2s[tid] = w2[tid];
    wls[tid] = w1[128 * DIM + tid];          // length-feature row of w1
  }
  __syncthreads();

  const float b2v = b2[0];
  const long long wave_id = (long long)blockIdx.x * 8 + wv;

  for (long long g = wave_id; g < EDGE_GROUPS; g += EDGE_WAVES) {
    const long long e0 = g * 16;

    // lanes 0..15 own edge m=lane; values broadcast to the wave via shuffles
    int s_m = 0, d_m = 0;
    float len_m = 0.f;
    if (lane < 16) {
      s_m = (int)ei[e0 + lane];
      d_m = (int)ei[(long long)NE + e0 + lane];
      const float dx = pos[d_m * 3 + 0] - pos[s_m * 3 + 0];
      const float dy = pos[d_m * 3 + 1] - pos[s_m * 3 + 1];
      const float dz = pos[d_m * 3 + 2] - pos[s_m * 3 + 2];
      len_m = sqrtf(dx * dx + dy * dy + dz * dz);
    }
    const int mySrc = __shfl(s_m, m16, 32);   // row for this lane's A fragment
    const int myDst = __shfl(d_m, m16, 32);

    // gather A fragments: x[src] / x[dst] rows (L2-resident), two K-halves
    const float* xs = x + (long long)mySrc * DIM;
    const float* xd = x + (long long)myDst * DIM;
    const v16h a_s0 = load_a_frag(xs, 0, lane);
    const v16h a_s1 = load_a_frag(xs, 32, lane);
    const v16h a_d0 = load_a_frag(xd, 0, lane);
    const v16h a_d1 = load_a_frag(xd, 32, lane);

    // lengths for this half-wave's C rows (M = r + 8*half)
    float lenr[8];
#pragma unroll
    for (int r = 0; r < 8; ++r) lenr[r] = __shfl(len_m, r + 8 * half, 32);

    float pr[8];
#pragma unroll
    for (int r = 0; r < 8; ++r) pr[r] = 0.f;

#pragma unroll
    for (int t = 0; t < 4; ++t) {
      const int n = t * 16 + m16;
      const float b1n = b1s[n], wl = wls[n];
      v8f acc;
#pragma unroll
      for (int r = 0; r < 8; ++r) acc[r] = b1n + lenr[r] * wl;  // fp32 init
      v16h b;
      b = load_b_frag(&w1t[0][0], W1_STRIDE, t * 16, 0, lane);
      acc = __builtin_amdgcn_wmma_f32_16x16x32_f16(false, a_s0, false, b, (short)0, acc, false, false);
      b = load_b_frag(&w1t[0][0], W1_STRIDE, t * 16, 32, lane);
      acc = __builtin_amdgcn_wmma_f32_16x16x32_f16(false, a_s1, false, b, (short)0, acc, false, false);
      b = load_b_frag(&w1t[0][0], W1_STRIDE, t * 16, 64, lane);
      acc = __builtin_amdgcn_wmma_f32_16x16x32_f16(false, a_d0, false, b, (short)0, acc, false, false);
      b = load_b_frag(&w1t[0][0], W1_STRIDE, t * 16, 96, lane);
      acc = __builtin_amdgcn_wmma_f32_16x16x32_f16(false, a_d1, false, b, (short)0, acc, false, false);

      const float w2n = w2s[n];
#pragma unroll
      for (int r = 0; r < 8; ++r) pr[r] += fmaxf(acc[r], 0.f) * w2n;  // ReLU+dot
    }

    // butterfly-reduce h@w2 over the 16 N-lanes of each half-wave;
    // afterwards every lane holds the full row sum for M = r + 8*half
#pragma unroll
    for (int r = 0; r < 8; ++r) {
      pr[r] += __shfl_xor(pr[r], 1, 32);
      pr[r] += __shfl_xor(pr[r], 2, 32);
      pr[r] += __shfl_xor(pr[r], 4, 32);
      pr[r] += __shfl_xor(pr[r], 8, 32);
    }
    float ewreg[8];
#pragma unroll
    for (int r = 0; r < 8; ++r)
      ewreg[r] = 1.f / (1.f + __expf(-(pr[r] + b2v)));   // sigmoid
    if (m16 == 0) {                       // lane 0: rows 0..7, lane 16: rows 8..15
#pragma unroll
      for (int r = 0; r < 8; ++r) ew_out[e0 + r + 8 * half] = ewreg[r];
    }

    // scatter: out[dst] += x_t[src] * ew   (2 columns per lane, 16 edges)
    const int c0 = lane * 2;
#pragma unroll
    for (int m = 0; m < 16; ++m) {
      const float w  = __shfl(ewreg[m & 7], (m & 8) ? 16 : 0, 32);
      const int   sm = __shfl(s_m, m, 32);
      const int   dm = __shfl(d_m, m, 32);
      const float2 v = *(const float2*)(xt + (long long)sm * DIM + c0);
      float* o = out + (long long)dm * DIM + c0;
      atomic_add_f32(o + 0, v.x * w);
      atomic_add_f32(o + 1, v.y * w);
    }
  }
}

extern "C" void kernel_launch(void* const* d_in, const int* in_sizes, int n_in,
                              void* d_out, int out_size, void* d_ws, size_t ws_size,
                              hipStream_t stream) {
  const float*     x   = (const float*)d_in[0];
  const long long* ei  = (const long long*)d_in[1];   // int64 edge_index [2,E]
  const float*     pos = (const float*)d_in[2];
  const float*     w1  = (const float*)d_in[3];
  const float*     b1  = (const float*)d_in[4];
  const float*     w2  = (const float*)d_in[5];
  const float*     b2  = (const float*)d_in[6];
  const float*     wt  = (const float*)d_in[7];
  const float*     bt  = (const float*)d_in[8];

  float* out    = (float*)d_out;                        // [N,64]
  float* ew_out = out + (long long)NN * DIM;            // [E]
  float* xt     = (float*)d_ws;                         // [N,64] scratch (25.6 MB)

  // 1) zero accumulation region (d_out is poisoned by the harness)
  const long long n4 = (long long)NN * DIM / 4;
  wec_zero_kernel<<<(int)((n4 + 255) / 256), 256, 0, stream>>>((float4*)out, n4);

  // 2) node transform x_t = x @ wt + bt  (WMMA, 16 rows/wave)
  wec_node_transform_kernel<<<(NN / 16 + 7) / 8, 256, 0, stream>>>(x, wt, bt, xt);

  // 3) fused edge MLP + sigmoid + weighted scatter-add (WMMA, 10x16 edges/wave)
  wec_edge_mlp_scatter_kernel<<<EDGE_BLOCKS, 256, 0, stream>>>(
      x, ei, pos, w1, b1, w2, b2, xt, out, ew_out);
}
